// GraphConvolutionwithDephSep_32976758899296
// MI455X (gfx1250) — matrix operand order repaired and checked by reading, not compiled, not run on hardware
//
#include <hip/hip_runtime.h>

typedef __attribute__((ext_vector_type(16))) __bf16 v16bf;
typedef __attribute__((ext_vector_type(8)))  __bf16 v8bf;
typedef __attribute__((ext_vector_type(8)))  float  v8f;
typedef __attribute__((ext_vector_type(4)))  float  v4f;

#define NROWS 8192
#define CH    128
#define BM    128      // rows per block (8 waves x 16 rows)
#define WAVES 8
#define KT    32       // WMMA K per step (bf16 16x16x32)
#define TILE_ELEMS (3 * CH * KT)          // 12288 bf16 = 24 KB per buffer

// ---------------------------------------------------------------------------
// Prologue 1: xT[s][c][k] = bf16( x[k][c] * sd_s[c] )   (3 x 128 x 8192 bf16)
// (S@x)*sd == S@(x*sd): all three support-GEMMs share one f32 accumulator.
// ---------------------------------------------------------------------------
__global__ void prep_xT(const float* __restrict__ x,
                        const float* __restrict__ sd0,
                        const float* __restrict__ sd1,
                        const float* __restrict__ sd2,
                        __bf16* __restrict__ xT)
{
    int j = blockIdx.x * blockDim.x + threadIdx.x;
    if (j >= NROWS) return;
    for (int c = 0; c < CH; ++c) {
        float v = x[(size_t)j * CH + c];
        xT[((size_t)0 * CH + c) * NROWS + j] = (__bf16)(v * sd0[c]);
        xT[((size_t)1 * CH + c) * NROWS + j] = (__bf16)(v * sd1[c]);
        xT[((size_t)2 * CH + c) * NROWS + j] = (__bf16)(v * sd2[c]);
    }
}

// Prologue 2: WT[o][c] = bf16( W[c][o] )
__global__ void prep_WT(const float* __restrict__ W, __bf16* __restrict__ WT)
{
    int o = blockIdx.x, c = threadIdx.x;
    WT[(size_t)o * CH + c] = (__bf16)W[(size_t)c * CH + o];
}

// ---------------------------------------------------------------------------
// Fused: agg = s0@xT0 + s1@xT1 + s2@xT2 with bf16 WMMA / f32 acc.
// B tiles staged in LDS via async global->LDS DMA, double-buffered per k-tile;
// A streamed HBM->VGPR (cvt to bf16), one-tile register double buffer.
// Epilogue: out = relu(agg @ W) in-block via LDS + 4 WMMA steps.
// ---------------------------------------------------------------------------
__global__ __launch_bounds__(WAVES * 32) void gcn_fused(
    const float*  __restrict__ s0,
    const float*  __restrict__ s1,
    const float*  __restrict__ s2,
    const __bf16* __restrict__ xT,   // [3][CH][NROWS] bf16
    const __bf16* __restrict__ WT,   // [CH][CH] bf16 (out-major)
    float*        __restrict__ out)  // [NROWS][CH] f32
{
    __shared__ __align__(64) __bf16 btile[2][TILE_ELEMS];   // 2 x 24 KB
    __shared__ __align__(32) __bf16 lds_agg[WAVES][16 * CH]; // 32 KB

    const int tid  = threadIdx.x;
    const int lane = tid & 31;
    const int wave = tid >> 5;
    const int half = lane >> 4;    // 0: lanes 0-15, 1: lanes 16-31
    const int l16  = lane & 15;
    const int rowBase = blockIdx.x * BM + wave * 16;

    // A layout (contiguous k order): lane-half 0 holds K {0..15}, half 1 {16..31}
    const int aK0 = half * 16;
    const int aK1 = half * 16 + 8;

    const float* __restrict__ sA[3] = {
        s0 + (size_t)(rowBase + l16) * NROWS,
        s1 + (size_t)(rowBase + l16) * NROWS,
        s2 + (size_t)(rowBase + l16) * NROWS };

    const unsigned ldsBase[2] = {
        (unsigned)(uintptr_t)&btile[0][0],
        (unsigned)(uintptr_t)&btile[1][0] };

    // ---- async fill of one B tile: 1536 x 16B units, 6 per thread ----------
    // LDS layout: row = s*128+c owns 64B (32 k values); its four 16B units are
    // rotated by rot(row) = (row + row>>2) & 3 for bank-conflict-free reads.
    auto fillTile = [&](int ktL, int p) {
#pragma unroll
        for (int i = 0; i < 6; ++i) {
            int e   = tid + i * 256;
            int row = e >> 2;
            int u   = e & 3;
            int rot = (row + (row >> 2)) & 3;
            unsigned ldsOff = ldsBase[p] + (unsigned)(row * 64 + ((u + rot) & 3) * 16);
            const char* g = (const char*)xT +
                            (size_t)row * (NROWS * 2) + (size_t)u * 16 + (size_t)ktL * 2;
            asm volatile("global_load_async_to_lds_b128 %0, %1, off"
                         :: "v"(ldsOff), "v"(g) : "memory");
        }
    };

    // ---- A fragment: 16 fp32 from one support row -> bf16 ------------------
    auto loadA = [&](int ktL, int s, v16bf& aF) {
        const float* ap = sA[s] + ktL;
        __builtin_prefetch(ap + 4 * KT, 0, 1);   // pull support stream HBM->L2 ahead
        v4f t0 = *(const v4f*)(ap + aK0);
        v4f t1 = *(const v4f*)(ap + aK0 + 4);
        v4f t2 = *(const v4f*)(ap + aK1);
        v4f t3 = *(const v4f*)(ap + aK1 + 4);
#pragma unroll
        for (int i = 0; i < 4; ++i) {
            aF[i]      = (__bf16)t0[i];
            aF[4 + i]  = (__bf16)t1[i];
            aF[8 + i]  = (__bf16)t2[i];
            aF[12 + i] = (__bf16)t3[i];
        }
    };

    // ---- B fragment from LDS tile (two swizzled 16B halves) ----------------
    const int rotL = (l16 + (l16 >> 2)) & 3;     // == rot(row) for our rows
    auto readB = [&](int p, int s, int n) -> v16bf {
        const __bf16* base = &btile[p][0] + (size_t)(s * CH + n * 16 + l16) * KT;
        int u0 = half * 2;
        v8bf lo = *(const v8bf*)(base + ((u0 + rotL) & 3) * 8);
        v8bf hi = *(const v8bf*)(base + ((u0 + 1 + rotL) & 3) * 8);
        v16bf b;
#pragma unroll
        for (int i = 0; i < 8; ++i) { b[i] = lo[i]; b[8 + i] = hi[i]; }
        return b;
    };

    v8f acc[8];
#pragma unroll
    for (int n = 0; n < 8; ++n) { v8f z = {}; acc[n] = z; }

    v16bf aB[2][3];

    // pipeline prologue: tile 0 in flight, A(kt=0) in registers
    fillTile(0, 0);
#pragma unroll
    for (int s = 0; s < 3; ++s) loadA(0, s, aB[0][s]);

    for (int kt = 0; kt < NROWS; kt += 2 * KT) {
#pragma unroll
        for (int u = 0; u < 2; ++u) {
            const int p = u;          // current LDS buffer / A register parity
            const int q = p ^ 1;
            const int ktCur = kt + u * KT;
            const int ktNxt = ktCur + KT;
            const bool hasNext = (ktNxt < NROWS);
            if (hasNext) {
                fillTile(ktNxt, q);
#pragma unroll
                for (int s = 0; s < 3; ++s) loadA(ktNxt, s, aB[q][s]);
                asm volatile("s_wait_asynccnt 0x6" ::: "memory");
            } else {
                asm volatile("s_wait_asynccnt 0x0" ::: "memory");
            }
            __syncthreads();          // tile[p] visible to all waves
#pragma unroll
            for (int s = 0; s < 3; ++s)
#pragma unroll
                for (int n = 0; n < 8; ++n)
                    acc[n] = __builtin_amdgcn_wmma_f32_16x16x32_bf16(
                        false, aB[p][s], false, readB(p, s, n), (short)0,
                        acc[n], false, false);
            __syncthreads();          // tile[p] free for refill
        }
    }

    // ---- epilogue: agg tile (16x128) -> LDS as bf16, wave-private region ----
    __bf16* aggw = &lds_agg[wave][0];
#pragma unroll
    for (int n = 0; n < 8; ++n)
#pragma unroll
        for (int r = 0; r < 8; ++r)
            aggw[(r + half * 8) * CH + n * 16 + l16] = (__bf16)acc[n][r];

    __syncthreads();

    // ---- projection: out(16x128) = relu( agg(16x128) @ W(128x128) ) ----
    v8f oacc[8];
#pragma unroll
    for (int n = 0; n < 8; ++n) { v8f z = {}; oacc[n] = z; }

    const __bf16* arow = aggw + l16 * CH;
#pragma unroll
    for (int k2 = 0; k2 < CH; k2 += KT) {
        v8bf lo = *(const v8bf*)(arow + k2 + aK0);
        v8bf hi = *(const v8bf*)(arow + k2 + aK1);
        v16bf a;
#pragma unroll
        for (int i = 0; i < 8; ++i) { a[i] = lo[i]; a[8 + i] = hi[i]; }
#pragma unroll
        for (int n = 0; n < 8; ++n) {
            const __bf16* bp = WT + (size_t)(n * 16 + l16) * CH + k2 + half * 16;
            v16bf b = *(const v16bf*)bp;
            oacc[n] = __builtin_amdgcn_wmma_f32_16x16x32_bf16(
                false, a, false, b, (short)0, oacc[n], false, false);
        }
    }

    // relu + store (C/D layout: VGPR r -> row r + half*8, lane -> column)
#pragma unroll
    for (int n = 0; n < 8; ++n)
#pragma unroll
        for (int r = 0; r < 8; ++r) {
            int row = rowBase + r + half * 8;
            int col = n * 16 + l16;
            float v = oacc[n][r];
            out[(size_t)row * CH + col] = v > 0.f ? v : 0.f;
        }
}

// ---------------------------------------------------------------------------
extern "C" void kernel_launch(void* const* d_in, const int* in_sizes, int n_in,
                              void* d_out, int out_size, void* d_ws, size_t ws_size,
                              hipStream_t stream)
{
    (void)in_sizes; (void)n_in; (void)out_size; (void)ws_size;
    const float* x   = (const float*)d_in[0];
    const float* s0  = (const float*)d_in[1];
    const float* s1  = (const float*)d_in[2];
    const float* s2  = (const float*)d_in[3];
    const float* W   = (const float*)d_in[4];
    const float* sd0 = (const float*)d_in[5];
    const float* sd1 = (const float*)d_in[6];
    const float* sd2 = (const float*)d_in[7];
    float* out = (float*)d_out;

    __bf16* xT = (__bf16*)d_ws;                        // 3*128*8192*2B = 6 MB
    __bf16* WT = xT + (size_t)3 * CH * NROWS;          // +32 KB

    prep_xT<<<(NROWS + 255) / 256, 256, 0, stream>>>(x, sd0, sd1, sd2, xT);
    prep_WT<<<CH, CH, 0, stream>>>(W, WT);
    gcn_fused<<<NROWS / BM, WAVES * 32, 0, stream>>>(s0, s1, s2, xT, WT, out);
}
